// DeformableConv2d_10058813407827
// MI455X (gfx1250) — matrix operand order repaired and checked by reading, not compiled, not run on hardware
//
#include <hip/hip_runtime.h>

// ---------------------------------------------------------------- constants
#define BB   8
#define CIN  128
#define COUT 256
#define HH   56
#define WW   56
#define HWSZ (HH*WW)            // 3136
#define K2   9
#define NOFF 18
#define NOUT 27                 // 18 offset + 9 mask channels
#define MDIM (BB*HWSZ)          // 25088 = 196*128
#define KDIM (K2*CIN)           // 1152, ordered Kidx = k*128 + c

typedef __attribute__((ext_vector_type(8)))  __bf16 v8bf;
typedef __attribute__((ext_vector_type(16))) __bf16 v16bf;
typedef __attribute__((ext_vector_type(8)))  float  v8f;
typedef __attribute__((__vector_size__(16))) int    v4i_g;   // matches builtin param type

// ---------------------------------------------------------------- helpers
__device__ __forceinline__ unsigned short f2bf(float f) {
  unsigned u = __float_as_uint(f);
  unsigned r = u + 0x7FFFu + ((u >> 16) & 1u);   // round-to-nearest-even
  return (unsigned short)(r >> 16);
}

#if __has_builtin(__builtin_amdgcn_global_load_async_to_lds_b128)
#define ASYNC_LDS 1
#else
#define ASYNC_LDS 0
#endif

// copy 32 bytes (16 bf16) global -> LDS
__device__ __forceinline__ void stage32(const unsigned short* g, unsigned short* l) {
#if ASYNC_LDS
  __builtin_amdgcn_global_load_async_to_lds_b128((v4i_g*)g,       (v4i_g*)l,       0, 0);
  __builtin_amdgcn_global_load_async_to_lds_b128((v4i_g*)(g + 8), (v4i_g*)(l + 8), 0, 0);
#else
  const uint4* gs = (const uint4*)g;
  uint4* ls = (uint4*)l;
  ls[0] = gs[0];
  ls[1] = gs[1];
#endif
}

__device__ __forceinline__ void wait_stage() {
#if ASYNC_LDS
 #if __has_builtin(__builtin_amdgcn_s_wait_asynccnt)
  __builtin_amdgcn_s_wait_asynccnt(0);
 #else
  asm volatile("s_wait_asynccnt 0x0" ::: "memory");
 #endif
#endif
}

// A-frag (16x32 bf16): lane&15 = row M, lane>>4 selects K halves {0..7,16..23} vs {8..15,24..31}
__device__ __forceinline__ v16bf frag_a(const unsigned short* rowbase, int half) {
  v8bf lo = *(const v8bf*)(rowbase + half * 8);
  v8bf hi = *(const v8bf*)(rowbase + 16 + half * 8);
  return __builtin_shufflevector(lo, hi, 0,1,2,3,4,5,6,7,8,9,10,11,12,13,14,15);
}
// B-frag (32x16 bf16), LDS stored as [N][K]: lane&15 = col N, lane>>4 selects K halves {0..7,16..23} vs {8..15,24..31}
__device__ __forceinline__ v16bf frag_b(const unsigned short* rowbase, int half) {
  v8bf lo = *(const v8bf*)(rowbase + half * 8);
  v8bf hi = *(const v8bf*)(rowbase + 16 + half * 8);
  return __builtin_shufflevector(lo, hi, 0,1,2,3,4,5,6,7,8,9,10,11,12,13,14,15);
}

// ======================================================== kernel 1: offsets+mask
__global__ __launch_bounds__(256) void offmask_kernel(
    const float* __restrict__ x,  const float* __restrict__ w_off,
    const float* __restrict__ b_off, const float* __restrict__ w_mask,
    const float* __restrict__ b_mask,
    float* __restrict__ py, float* __restrict__ px, float* __restrict__ msk)
{
  __shared__ __align__(16) float wsh[32 * NOUT * 9];   // 31104 B
  const int tid = threadIdx.x;
  const int pos = blockIdx.x * 256 + tid;              // 98*256 == MDIM exact
  const int b  = pos / HWSZ, hw = pos % HWSZ;
  const int h  = hw / WW,  wo = hw % WW;

  float acc[NOUT];
#pragma unroll
  for (int j = 0; j < NOFF; ++j) acc[j] = b_off[j];
#pragma unroll
  for (int j = 0; j < K2; ++j)   acc[NOFF + j] = b_mask[j];

  for (int cc = 0; cc < CIN; cc += 32) {
    __syncthreads();
    for (int i = tid; i < 32 * NOUT * 9; i += 256) {
      int cl = i / (NOUT * 9), rem = i % (NOUT * 9);
      int j = rem / 9, t = rem % 9;
      int c = cc + cl;
      wsh[i] = (j < NOFF) ? w_off[(j * CIN + c) * 9 + t]
                          : w_mask[((j - NOFF) * CIN + c) * 9 + t];
    }
    __syncthreads();
    for (int cl = 0; cl < 32; ++cl) {
      const float* xc = x + ((size_t)(b * CIN + cc + cl)) * HWSZ;
      float xv[9];
#pragma unroll
      for (int t = 0; t < 9; ++t) {
        int yy = h - 1 + t / 3, xx = wo - 1 + t % 3;
        bool ok = (yy >= 0) & (yy < HH) & (xx >= 0) & (xx < WW);
        xv[t] = ok ? xc[yy * WW + xx] : 0.f;
      }
      const float* wr = &wsh[cl * NOUT * 9];
#pragma unroll
      for (int j = 0; j < NOUT; ++j) {
#pragma unroll
        for (int t = 0; t < 9; ++t)
          acc[j] = fmaf(wr[j * 9 + t], xv[t], acc[j]);
      }
    }
  }
#pragma unroll
  for (int k = 0; k < K2; ++k) {
    float mk = 1.f / (1.f + __expf(-acc[NOFF + k]));
    int off = (b * K2 + k) * HWSZ + hw;
    py[off]  = (float)(h  - 1 + k / 3) + acc[2 * k];
    px[off]  = (float)(wo - 1 + k % 3) + acc[2 * k + 1];
    msk[off] = mk;
  }
}

// ======================================================== kernel 2: weight repack
__global__ __launch_bounds__(256) void wprep_kernel(const float* __restrict__ w,
                                                    unsigned short* __restrict__ Wb)
{
  int i = blockIdx.x * 256 + threadIdx.x;     // 1152*256 == COUT*KDIM exact
  int o = i / KDIM, r = i % KDIM;
  int k = r / CIN, c = r % CIN;
  Wb[i] = f2bf(w[(o * CIN + c) * 9 + k]);
}

// ======================================================== kernel 3: bilinear sample
__global__ __launch_bounds__(256) void sample_kernel(
    const float* __restrict__ x, const float* __restrict__ py,
    const float* __restrict__ px, const float* __restrict__ msk,
    unsigned short* __restrict__ A)
{
  int idx = blockIdx.x * 256 + threadIdx.x;   // 882*256 == MDIM*K2 exact
  int pos = idx / K2, k = idx % K2;
  int b = pos / HWSZ, hw = pos % HWSZ;
  int off = (b * K2 + k) * HWSZ + hw;

  float pyv = py[off], pxv = px[off], mv = msk[off];
  float y0f = floorf(pyv), x0f = floorf(pxv);
  float wy1 = pyv - y0f,  wx1 = pxv - x0f;
  float wy[2] = {1.f - wy1, wy1};
  float wx[2] = {1.f - wx1, wx1};
  int y0 = (int)y0f, x0 = (int)x0f;

  float wgt[4];
  int   cidx[4];
#pragma unroll
  for (int iy = 0; iy < 2; ++iy)
#pragma unroll
    for (int ix = 0; ix < 2; ++ix) {
      int yi = y0 + iy, xi = x0 + ix;
      bool ok = (yi >= 0) & (yi < HH) & (xi >= 0) & (xi < WW);
      int yc = min(max(yi, 0), HH - 1), xc = min(max(xi, 0), WW - 1);
      wgt[iy * 2 + ix]  = ok ? wy[iy] * wx[ix] : 0.f;
      cidx[iy * 2 + ix] = yc * WW + xc;
    }

  const float* xb = x + (size_t)b * CIN * HWSZ;
  unsigned short* Arow = A + (size_t)pos * KDIM + k * CIN;
  for (int c = 0; c < CIN; ++c) {
    const float* xc = xb + c * HWSZ;
    float v = wgt[0] * xc[cidx[0]] + wgt[1] * xc[cidx[1]]
            + wgt[2] * xc[cidx[2]] + wgt[3] * xc[cidx[3]];
    Arow[c] = f2bf(v * mv);
  }
}

// ======================================================== kernel 4: WMMA GEMM
// out[M=25088][N=256] = A[M][K=1152] * Wb[N][K]^T + bias, scattered to NCHW
__global__ __launch_bounds__(256) void gemm_kernel(
    const unsigned short* __restrict__ A, const unsigned short* __restrict__ Wb,
    const float* __restrict__ bias, float* __restrict__ out)
{
  __shared__ __align__(16) unsigned short As[128 * 32];   // [m][k]
  __shared__ __align__(16) unsigned short Bs[128 * 32];   // [n][k]

  const int tid  = threadIdx.x;
  const int lane = tid & 31, wave = tid >> 5;
  const int wm = wave >> 1;            // 0..3 -> 32-row slab
  const int wn = wave & 1;             // 0..1 -> 64-col slab
  const int mbase = blockIdx.x * 128;  // 196 blocks, exact
  const int nbase = blockIdx.y * 128;  // 2 blocks, exact

  const v8f vzero = {0.f,0.f,0.f,0.f,0.f,0.f,0.f,0.f};
  v8f acc[2][4];
#pragma unroll
  for (int i = 0; i < 2; ++i)
#pragma unroll
    for (int j = 0; j < 4; ++j) acc[i][j] = vzero;

  // staging: each of 256 threads moves 32B of A and 32B of B per k-step
  const int srow = tid >> 1;
  const int sseg = (tid & 1) * 16;
  const unsigned short* Ag = A  + (size_t)(mbase + srow) * KDIM + sseg;
  const unsigned short* Bg = Wb + (size_t)(nbase + srow) * KDIM + sseg;
  unsigned short* Asl = &As[srow * 32 + sseg];
  unsigned short* Bsl = &Bs[srow * 32 + sseg];

  const int r = lane & 15, half = lane >> 4;
  const unsigned short* Abase = &As[(wm * 32 + r) * 32];
  const unsigned short* Bbase = &Bs[(wn * 64 + r) * 32];

  for (int kt = 0; kt < KDIM; kt += 32) {
    stage32(Ag + kt, Asl);
    stage32(Bg + kt, Bsl);
    wait_stage();
    __syncthreads();

    v16bf a0 = frag_a(Abase,           half);
    v16bf a1 = frag_a(Abase + 16 * 32, half);
    v16bf b0 = frag_b(Bbase,           half);
    v16bf b1 = frag_b(Bbase + 16 * 32, half);
    v16bf b2 = frag_b(Bbase + 32 * 32, half);
    v16bf b3 = frag_b(Bbase + 48 * 32, half);

    acc[0][0] = __builtin_amdgcn_wmma_f32_16x16x32_bf16(false, a0, false, b0, (short)0, acc[0][0], false, false);
    acc[0][1] = __builtin_amdgcn_wmma_f32_16x16x32_bf16(false, a0, false, b1, (short)0, acc[0][1], false, false);
    acc[0][2] = __builtin_amdgcn_wmma_f32_16x16x32_bf16(false, a0, false, b2, (short)0, acc[0][2], false, false);
    acc[0][3] = __builtin_amdgcn_wmma_f32_16x16x32_bf16(false, a0, false, b3, (short)0, acc[0][3], false, false);
    acc[1][0] = __builtin_amdgcn_wmma_f32_16x16x32_bf16(false, a1, false, b0, (short)0, acc[1][0], false, false);
    acc[1][1] = __builtin_amdgcn_wmma_f32_16x16x32_bf16(false, a1, false, b1, (short)0, acc[1][1], false, false);
    acc[1][2] = __builtin_amdgcn_wmma_f32_16x16x32_bf16(false, a1, false, b2, (short)0, acc[1][2], false, false);
    acc[1][3] = __builtin_amdgcn_wmma_f32_16x16x32_bf16(false, a1, false, b3, (short)0, acc[1][3], false, false);
    __syncthreads();
  }

  // epilogue: C/D layout — lane&15 = N, lane>>4 picks M {+0,+8}, vgpr v = M row
#pragma unroll
  for (int fm = 0; fm < 2; ++fm)
#pragma unroll
    for (int fn = 0; fn < 4; ++fn) {
      int n = nbase + wn * 64 + fn * 16 + r;
      float bv = bias[n];
#pragma unroll
      for (int v = 0; v < 8; ++v) {
        int m = mbase + wm * 32 + fm * 16 + half * 8 + v;
        int bimg = m / HWSZ, hw = m % HWSZ;
        out[((size_t)(bimg * COUT + n)) * HWSZ + hw] = acc[fm][fn][v] + bv;
      }
    }
}

// ======================================================== launch
extern "C" void kernel_launch(void* const* d_in, const int* in_sizes, int n_in,
                              void* d_out, int out_size, void* d_ws, size_t ws_size,
                              hipStream_t stream) {
  (void)in_sizes; (void)n_in; (void)out_size; (void)ws_size;
  const float* x      = (const float*)d_in[0];
  const float* w_off  = (const float*)d_in[1];
  const float* b_off  = (const float*)d_in[2];
  const float* w_mask = (const float*)d_in[3];
  const float* b_mask = (const float*)d_in[4];
  const float* w      = (const float*)d_in[5];
  const float* bias   = (const float*)d_in[6];
  float* out = (float*)d_out;

  char* ws = (char*)d_ws;
  size_t off = 0;
  auto alloc = [&](size_t bytes) {
    void* p = ws + off;
    off = (off + bytes + 255) & ~(size_t)255;
    return p;
  };
  float* py  = (float*)alloc((size_t)BB * K2 * HWSZ * sizeof(float));
  float* px  = (float*)alloc((size_t)BB * K2 * HWSZ * sizeof(float));
  float* msk = (float*)alloc((size_t)BB * K2 * HWSZ * sizeof(float));
  unsigned short* Wb = (unsigned short*)alloc((size_t)COUT * KDIM * 2);
  unsigned short* A  = (unsigned short*)alloc((size_t)MDIM * KDIM * 2);   // ~58 MB

  offmask_kernel<<<MDIM / 256, 256, 0, stream>>>(x, w_off, b_off, w_mask, b_mask, py, px, msk);
  wprep_kernel<<<(COUT * KDIM) / 256, 256, 0, stream>>>(w, Wb);
  sample_kernel<<<(MDIM * K2) / 256, 256, 0, stream>>>(x, py, px, msk, A);
  dim3 grid(MDIM / 128, COUT / 128);
  gemm_kernel<<<grid, 256, 0, stream>>>(A, Wb, bias, out);
}
